// ClusteringLayer_13786845020239
// MI455X (gfx1250) — matrix operand order repaired and checked by reading, not compiled, not run on hardware
//
#include <hip/hip_runtime.h>

typedef __attribute__((ext_vector_type(16))) __bf16 v16bf;
typedef __attribute__((ext_vector_type(8)))  __bf16 bfx8;
typedef __attribute__((ext_vector_type(4)))  __bf16 bfx4;
typedef __attribute__((ext_vector_type(8)))  float  v8f;

#define CDIM       256
#define KCL        512
#define M_TOTAL    131072          // 32*64*64 tokens
#define ROWS_WG    128
#define TPB        256
#define LSTRIDE    264             // padded bf16 row stride (528 B: bank-conflict-free, 16B aligned)

// LDS layout (bytes)
#define A_SZ       (ROWS_WG * LSTRIDE * 2)    // 67584
#define A_HI_OFF   0
#define A_LO_OFF   (A_HI_OFF + A_SZ)
#define B_SZ       (16 * LSTRIDE * 2)         // 8448
#define B_HI_OFF   (A_LO_OFF + A_SZ)
#define B_LO_OFF   (B_HI_OFF + B_SZ)
#define CSQ_OFF    (B_LO_OFF + B_SZ)
#define BEST_OFF   (CSQ_OFF + KCL * 4)
#define SMEM_BYTES (BEST_OFF + ROWS_WG * 4)   // ~151 KB

__device__ __forceinline__ void split_bf16(float f, __bf16& h, __bf16& l) {
  h = (__bf16)f;
  l = (__bf16)(f - (float)h);
}

// ---- tiny kernel: csq[k] = ||centers[k]||^2 ----
__global__ void csq_kernel(const float* __restrict__ centers, float* __restrict__ csq) {
  int k = blockIdx.x * blockDim.x + threadIdx.x;
  if (k < KCL) {
    const float4* c = (const float4*)(centers + (size_t)k * CDIM);
    float s = 0.f;
#pragma unroll 8
    for (int i = 0; i < CDIM / 4; ++i) {
      float4 v = c[i];
      s += v.x * v.x + v.y * v.y + v.z * v.z + v.w * v.w;
    }
    csq[k] = s;
  }
}

__global__ void kmeans_assign_kernel(const float* __restrict__ x,
                                     const float* __restrict__ centers,
                                     const float* __restrict__ csq,
                                     float* __restrict__ out_x,
                                     float* __restrict__ out_y) {
  extern __shared__ char smem[];
  __bf16* aHi   = (__bf16*)(smem + A_HI_OFF);
  __bf16* aLo   = (__bf16*)(smem + A_LO_OFF);
  __bf16* bHi   = (__bf16*)(smem + B_HI_OFF);
  __bf16* bLo   = (__bf16*)(smem + B_LO_OFF);
  float*  csqL  = (float*)(smem + CSQ_OFF);
  int*    bestL = (int*)(smem + BEST_OFF);

  const int t    = threadIdx.x;
  const int wg   = blockIdx.x;
  const int lane = t & 31;
  const int wave = t >> 5;
  const int half = lane >> 4;     // which 16-lane half of the wave32
  const int l16  = lane & 15;

  // ---- Phase 1: stage x tile as bf16 hi/lo into LDS; mirror x to output ----
  {
    const float4* x4 = (const float4*)x    + (size_t)wg * (ROWS_WG * CDIM / 4);
    float4*       o4 = (float4*)out_x      + (size_t)wg * (ROWS_WG * CDIM / 4);
#pragma unroll 4
    for (int i = 0; i < 32; ++i) {
      int u = i * TPB + t;            // float4 index within tile
      float4 v = x4[u];
      o4[u] = v;
      int row = u >> 6;               // 64 float4 per 256-elem row
      int col = (u & 63) * 4;
      bfx4 hv, lv;
      __bf16 h, l;
      split_bf16(v.x, h, l); hv[0] = h; lv[0] = l;
      split_bf16(v.y, h, l); hv[1] = h; lv[1] = l;
      split_bf16(v.z, h, l); hv[2] = h; lv[2] = l;
      split_bf16(v.w, h, l); hv[3] = h; lv[3] = l;
      *(bfx4*)(aHi + row * LSTRIDE + col) = hv;
      *(bfx4*)(aLo + row * LSTRIDE + col) = lv;
    }
    if (t < KCL / 4) {
      ((float4*)csqL)[t] = ((const float4*)csq)[t];
    }
  }
  __syncthreads();

  // per-lane running argmin state (8 rows owned by this lane in the C layout)
  float mind[8];
  int   mix[8];
#pragma unroll
  for (int v = 0; v < 8; ++v) { mind[v] = 3.4e38f; mix[v] = 0; }

  const __bf16* aHiRow = aHi + (wave * 16 + l16) * LSTRIDE;  // A: row = lane&15
  const __bf16* aLoRow = aLo + (wave * 16 + l16) * LSTRIDE;
  const __bf16* bHiCol = bHi + l16 * LSTRIDE;                // B: col = lane&15
  const __bf16* bLoCol = bLo + l16 * LSTRIDE;

  // ---- Phase 2: loop over 32 column tiles of 16 centers ----
  for (int ct = 0; ct < 32; ++ct) {
    const int c0 = ct * 16;
    // cooperative stage of 16 centers (bf16 hi/lo) into LDS
    {
      int brow = t >> 4;             // 0..15 center within tile
      int bcol = (t & 15) * 16;      // 16 elems per thread
      const float4* c4 = (const float4*)(centers + (size_t)(c0 + brow) * CDIM + bcol);
#pragma unroll
      for (int j = 0; j < 4; ++j) {
        float4 v = c4[j];
        bfx4 hv, lv;
        __bf16 h, l;
        split_bf16(v.x, h, l); hv[0] = h; lv[0] = l;
        split_bf16(v.y, h, l); hv[1] = h; lv[1] = l;
        split_bf16(v.z, h, l); hv[2] = h; lv[2] = l;
        split_bf16(v.w, h, l); hv[3] = h; lv[3] = l;
        *(bfx4*)(bHi + brow * LSTRIDE + bcol + j * 4) = hv;
        *(bfx4*)(bLo + brow * LSTRIDE + bcol + j * 4) = lv;
      }
    }
    __syncthreads();

    v8f acc = {};
#pragma unroll
    for (int kc = 0; kc < 8; ++kc) {
      const int k0 = kc * 32;
      union Frag { v16bf v; bfx8 q[2]; } ah, al, bh, bl;
      // A 16x32 bf16 layout: lane<16 -> K {k0..k0+7, k0+16..k0+23}; lane>=16 -> +8
      ah.q[0] = *(const bfx8*)(aHiRow + k0 + 8 * half);
      ah.q[1] = *(const bfx8*)(aHiRow + k0 + 16 + 8 * half);
      al.q[0] = *(const bfx8*)(aLoRow + k0 + 8 * half);
      al.q[1] = *(const bfx8*)(aLoRow + k0 + 16 + 8 * half);
      // B 32x16 bf16 layout: lane<16 -> K {k0..k0+7, k0+16..k0+23}; lane>=16 -> +8
      bh.q[0] = *(const bfx8*)(bHiCol + k0 + 8 * half);
      bh.q[1] = *(const bfx8*)(bHiCol + k0 + 16 + 8 * half);
      bl.q[0] = *(const bfx8*)(bLoCol + k0 + 8 * half);
      bl.q[1] = *(const bfx8*)(bLoCol + k0 + 16 + 8 * half);
      // bf16x3 split product: hi*hi + hi*lo + lo*hi, fp32 accumulate
      acc = __builtin_amdgcn_wmma_f32_16x16x32_bf16(false, ah.v, false, bh.v, (short)0, acc, false, false);
      acc = __builtin_amdgcn_wmma_f32_16x16x32_bf16(false, ah.v, false, bl.v, (short)0, acc, false, false);
      acc = __builtin_amdgcn_wmma_f32_16x16x32_bf16(false, al.v, false, bh.v, (short)0, acc, false, false);
    }

    const int col = c0 + l16;
    const float cs = csqL[col];
#pragma unroll
    for (int v = 0; v < 8; ++v) {
      float d = cs - 2.0f * acc[v];   // ||c||^2 - 2 f.c  (f_sq constant per row)
      if (d < mind[v]) { mind[v] = d; mix[v] = col; }
    }
    __syncthreads();                  // waves done with B tile before next overwrite
  }

  // ---- Phase 3: cross-lane argmin over the 16 columns per half ----
#pragma unroll
  for (int m = 8; m >= 1; m >>= 1) {
#pragma unroll
    for (int v = 0; v < 8; ++v) {
      float od = __shfl_xor(mind[v], m, 32);
      int   oi = __shfl_xor(mix[v],  m, 32);
      if (od < mind[v] || (od == mind[v] && oi < mix[v])) { mind[v] = od; mix[v] = oi; }
    }
  }
  if (l16 == 0) {
#pragma unroll
    for (int v = 0; v < 8; ++v)
      bestL[wave * 16 + half * 8 + v] = mix[v];  // C layout: row = v + 8*half
  }
  __syncthreads();

  // ---- Phase 4: gather winning centroids (coalesced float4) ----
#pragma unroll 4
  for (int i = 0; i < 32; ++i) {
    int rowL = i * 4 + (t >> 6);
    int c4i  = t & 63;
    int k = bestL[rowL];
    float4 v = ((const float4*)(centers + (size_t)k * CDIM))[c4i];
    ((float4*)out_y)[((size_t)wg * ROWS_WG + rowL) * 64 + c4i] = v;
  }
}

extern "C" void kernel_launch(void* const* d_in, const int* in_sizes, int n_in,
                              void* d_out, int out_size, void* d_ws, size_t ws_size,
                              hipStream_t stream) {
  const float* x       = (const float*)d_in[0];
  const float* centers = (const float*)d_in[1];
  float* csq   = (float*)d_ws;                         // 512 floats
  float* out_x = (float*)d_out;                        // x passthrough
  float* out_y = out_x + (size_t)M_TOTAL * CDIM;       // gathered centroids

  csq_kernel<<<2, 256, 0, stream>>>(centers, csq);
  kmeans_assign_kernel<<<M_TOTAL / ROWS_WG, TPB, SMEM_BYTES, stream>>>(
      x, centers, csq, out_x, out_y);
}
